// VectorOut_72808285602176
// MI455X (gfx1250) — matrix operand order repaired and checked by reading, not compiled, not run on hardware
//
#include <hip/hip_runtime.h>
#include <math.h>

typedef __attribute__((ext_vector_type(16))) _Float16 v16h;
typedef __attribute__((ext_vector_type(8)))  _Float16 v8h;
typedef __attribute__((ext_vector_type(4)))  _Float16 half4;
typedef __attribute__((ext_vector_type(8)))  float    v8f;

union V16 { v16h v; v8h h[2]; };

#define THREADS 128
#define TILE_ATOMS 64   // 4 waves x 16 atoms
#define RPAD 72         // padded row length (halves), 144B keeps 16B alignment

__global__ void __launch_bounds__(256)
zero_kernel(float* __restrict__ out, int n) {
    int t = blockIdx.x * blockDim.x + threadIdx.x;
    if (t < n) out[t] = 0.0f;
}

__global__ void __launch_bounds__(THREADS)
equiv_vec_kernel(const float* __restrict__ atom_out,   // [N,480]
                 const float* __restrict__ ln_w,       // [224]
                 const float* __restrict__ W1,         // [64,64] row-major (u,v)
                 const float* __restrict__ W2,         // [64,1]
                 const int*   __restrict__ batch_idx,  // [N]
                 float*       __restrict__ out,        // [B,3]
                 int N)
{
    // component-deinterleaved, row-major planes: plane[i][atom][u]
    __shared__ _Float16 plane[3][TILE_ATOMS][RPAD];    // 27.6 KB
    __shared__ _Float16 Wt[64][RPAD];                  // W1_1 transposed [v][u], 9.2 KB
    __shared__ float    normAcc[TILE_ATOMS];
    __shared__ float    invS[TILE_ATOMS];
    __shared__ float    wlnS[64];
    __shared__ float    w2S[64];
    __shared__ float    atomRes[TILE_ATOMS][3];

    const int t    = threadIdx.x;
    const int base = blockIdx.x * TILE_ATOMS;

    if (t < 64) {
        normAcc[t] = 0.0f;
        wlnS[t]    = ln_w[128 + t];   // l=1 affine weights
        w2S[t]     = W2[t];
    }
    // stage W1_1 transposed: packed b64 writes, L2-cached scattered reads
    for (int q = t; q < 64 * 16; q += THREADS) {       // 8 iters
        int v = q / 16, u0 = (q % 16) * 4;
        half4 p;
        #pragma unroll
        for (int j = 0; j < 4; ++j) p[j] = (_Float16)W1[(u0 + j) * 64 + v];
        *(half4*)&Wt[v][u0] = p;
    }
    __syncthreads();

    // ---- load: 12-float chunks (4 channels x 3 comps), deinterleave, norm ----
    for (int q = t; q < TILE_ATOMS * 16; q += THREADS) {  // 8 iters
        int a  = q / 16;
        int u0 = (q % 16) * 4;
        int ga = base + a;
        float f[12];
        if (ga < N) {
            const float* p = atom_out + (size_t)ga * 480 + 128 + (q % 16) * 12;
            float4 v0 = *(const float4*)(p);
            float4 v1 = *(const float4*)(p + 4);
            float4 v2 = *(const float4*)(p + 8);
            f[0]=v0.x; f[1]=v0.y; f[2]=v0.z; f[3]=v0.w;
            f[4]=v1.x; f[5]=v1.y; f[6]=v1.z; f[7]=v1.w;
            f[8]=v2.x; f[9]=v2.y; f[10]=v2.z; f[11]=v2.w;
        } else {
            #pragma unroll
            for (int k = 0; k < 12; ++k) f[k] = 0.0f;
        }
        float ss = 0.0f;
        #pragma unroll
        for (int k = 0; k < 12; ++k) ss += f[k] * f[k];
        atomicAdd(&normAcc[a], ss);                     // ds_add_f32
        #pragma unroll
        for (int i = 0; i < 3; ++i) {
            half4 p;
            #pragma unroll
            for (int j = 0; j < 4; ++j) p[j] = (_Float16)f[j * 3 + i];
            *(half4*)&plane[i][a][u0] = p;              // ds_store_b64
        }
    }
    __syncthreads();
    if (t < 64)
        invS[t] = __builtin_amdgcn_rsqf(normAcc[t] * (1.0f / 64.0f) + 1e-5f);
    __syncthreads();

    // ---- scale in place (vectorized 8 halves / op) ----
    for (int q = t; q < 3 * TILE_ATOMS * 8; q += THREADS) {  // 12 iters
        int i  = q / (TILE_ATOMS * 8);
        int r  = q % (TILE_ATOMS * 8);
        int a  = r / 8;
        int u0 = (r % 8) * 8;
        v8h x = *(v8h*)&plane[i][a][u0];                // ds_load_b128
        float inva = invS[a];
        #pragma unroll
        for (int e = 0; e < 8; ++e)
            x[e] = (_Float16)((float)x[e] * inva * wlnS[u0 + e]);
        *(v8h*)&plane[i][a][u0] = x;                    // ds_store_b128
    }
    __syncthreads();

    // ---- WMMA: per wave, 16 atoms; out[v,i] = sum_u A[m,u,i] * W[u,v] ----
    const int wave    = t >> 5;
    const int lane    = t & 31;
    const int laneHi  = lane >> 4;
    const int lmod    = lane & 15;
    const int atomRow = wave * 16 + lmod;   // A-matrix row (lanes 0-15 == 16-31)

    // A fragments: 16x32 f16 layout = two contiguous 8-half runs per lane
    V16 aF[3][2];
    #pragma unroll
    for (int i = 0; i < 3; ++i)
        #pragma unroll
        for (int kt = 0; kt < 2; ++kt) {
            const _Float16* rowp = &plane[i][atomRow][kt * 32 + laneHi * 8];
            aF[i][kt].h[0] = *(const v8h*)(rowp);        // K kt*32+8hi+0..7
            aF[i][kt].h[1] = *(const v8h*)(rowp + 16);   // K kt*32+8hi+16..23
        }

    float part[3][8];
    #pragma unroll
    for (int i = 0; i < 3; ++i)
        #pragma unroll
        for (int r = 0; r < 8; ++r) part[i][r] = 0.0f;

    #pragma unroll
    for (int nt = 0; nt < 4; ++nt) {
        // B fragment: two 8-K runs (8hi+0..7, 8hi+16..23) at fixed n -> contiguous in Wt row
        int n = nt * 16 + lmod;
        V16 b0, b1;
        {
            const _Float16* wr = &Wt[n][laneHi * 8];
            b0.h[0] = *(const v8h*)(wr);
            b0.h[1] = *(const v8h*)(wr + 16);
            b1.h[0] = *(const v8h*)(wr + 32);
            b1.h[1] = *(const v8h*)(wr + 48);
        }
        v8f acc[3];
        #pragma unroll
        for (int i = 0; i < 3; ++i) {
            v8f c = {};
            c = __builtin_amdgcn_wmma_f32_16x16x32_f16(false, aF[i][0].v, false, b0.v,
                                                       (short)0, c, false, false);
            c = __builtin_amdgcn_wmma_f32_16x16x32_f16(false, aF[i][1].v, false, b1.v,
                                                       (short)0, c, false, false);
            acc[i] = c;
        }
        // gate + W2 contraction; raw acc = f' * 8  (f' = acc/8)
        // args are O(1) and >= 1e-12: use raw v_sqrt/v_exp/v_rcp (no libm fixups)
        float w2v = w2S[n];
        #pragma unroll
        for (int r = 0; r < 8; ++r) {
            float fx = acc[0][r], fy = acc[1][r], fz = acc[2][r];
            float nrm = __builtin_amdgcn_sqrtf(
                (fx*fx + fy*fy + fz*fz) * (1.0f / 64.0f) + 1e-12f);
            float sig = __builtin_amdgcn_rcpf(1.0f + __expf(-nrm));
            float s   = sig * w2v * (1.0f / 64.0f);   // (1/8 linear) * (1/8 final)
            part[0][r] += fx * s;
            part[1][r] += fy * s;
            part[2][r] += fz * s;
        }
    }

    // ---- reduce over v across the 16 lanes of each half-wave ----
    #pragma unroll
    for (int i = 0; i < 3; ++i)
        #pragma unroll
        for (int r = 0; r < 8; ++r) {
            float v = part[i][r];
            v += __shfl_xor(v, 1, 32);
            v += __shfl_xor(v, 2, 32);
            v += __shfl_xor(v, 4, 32);
            v += __shfl_xor(v, 8, 32);
            if (lmod == 0)
                atomRes[wave * 16 + laneHi * 8 + r][i] = v;  // D row M=r(+8)
        }
    __syncthreads();

    // ---- segment sum: batch_idx sorted -> uniform-batch fast path ----
    int lastIdx = base + TILE_ATOMS - 1;
    if (lastIdx >= N) lastIdx = N - 1;
    bool uniform = (batch_idx[base] == batch_idx[lastIdx]);

    if (uniform) {
        if (t < 3) {
            int i = (t + 2) % 3;          // out col t <- component (t+2)%3  ((y,z,x)->(x,y,z))
            float s = 0.0f;
            #pragma unroll 8
            for (int a = 0; a < TILE_ATOMS; ++a) s += atomRes[a][i];
            unsafeAtomicAdd(&out[batch_idx[base] * 3 + t], s);   // global_atomic_add_f32
        }
    } else {
        if (t < TILE_ATOMS * 3) {
            int a = t / 3, i = t % 3;
            int ga = base + a;
            if (ga < N)
                unsafeAtomicAdd(&out[batch_idx[ga] * 3 + (i + 1) % 3], atomRes[a][i]);
        }
    }
}

extern "C" void kernel_launch(void* const* d_in, const int* in_sizes, int n_in,
                              void* d_out, int out_size, void* d_ws, size_t ws_size,
                              hipStream_t stream) {
    // setup_inputs order:
    // 0 atom_out, 1 coords, 2 ln_w, 3 ln_b, 4 W1_0, 5 W1_1, 6 W1_2, 7 W2,
    // 8 at_no, 9 batch_idx, 10 num_graphs
    const float* atom_out  = (const float*)d_in[0];
    const float* ln_w      = (const float*)d_in[2];
    const float* W1_1      = (const float*)d_in[5];
    const float* W2        = (const float*)d_in[7];
    const int*   batch_idx = (const int*)d_in[9];
    float* out = (float*)d_out;

    int N = in_sizes[0] / 480;

    zero_kernel<<<(out_size + 255) / 256, 256, 0, stream>>>(out, out_size);

    int blocks = (N + TILE_ATOMS - 1) / TILE_ATOMS;
    equiv_vec_kernel<<<blocks, THREADS, 0, stream>>>(atom_out, ln_w, W1_1, W2,
                                                     batch_idx, out, N);
}
